// Self_Attention_opt_v1_34978213659341
// MI455X (gfx1250) — matrix operand order repaired and checked by reading, not compiled, not run on hardware
//
#include <hip/hip_runtime.h>
#include <hip/hip_bf16.h>

// ---------------------------------------------------------------------------
// Self-attention (single head, d=1024) for MI455X / gfx1250.
//   B=4, S=2048, H=1024.  All matmuls: V_WMMA_F32_16X16X32_BF16.
// Pipeline:
//   k0a: X  -> Xb  (bf16, row major)           [one-shot convert, ~50MB traffic]
//   k0b: W  -> Wt  (bf16, TRANSPOSED [n][k])   [so GEMM B-tiles are TDM-able]
//   k1 : qkv GEMM, TDM double-buffered LDS staging, writes Q,K row-major bf16
//        and V transposed [b][h][s] bf16.
//   k2 : flash-style streaming-softmax attention, Q tile TDM-staged, fp32 out.
// ---------------------------------------------------------------------------

#define HID   1024
#define SEQ   2048
#define NBAT  4
#define N3    (3 * HID)

typedef __attribute__((ext_vector_type(16))) __bf16        v16bf;
typedef __attribute__((ext_vector_type(8)))  float         v8f;
typedef __attribute__((ext_vector_type(8)))  unsigned int  v8u;
typedef __attribute__((ext_vector_type(4)))  unsigned int  v4u;
typedef __attribute__((ext_vector_type(8)))  int           v8i;
typedef __attribute__((ext_vector_type(4)))  int           v4i;

#if __has_builtin(__builtin_amdgcn_tensor_load_to_lds) && \
    __has_builtin(__builtin_amdgcn_s_wait_tensorcnt)
#define HAVE_TDM 1
#else
#define HAVE_TDM 0
#endif

static __device__ __forceinline__ unsigned short f2bf(float f) {
    // round-to-nearest-even fp32 -> bf16
    unsigned int u = __builtin_bit_cast(unsigned int, f);
    unsigned int r = u + 0x7FFFu + ((u >> 16) & 1u);
    return (unsigned short)(r >> 16);
}
static __device__ __forceinline__ unsigned int pack2(unsigned short lo, unsigned short hi) {
    return (unsigned int)lo | ((unsigned int)hi << 16);
}
static __device__ __forceinline__ v16bf mk_frag(uint4 a, uint4 b) {
    v8u u = {a.x, a.y, a.z, a.w, b.x, b.y, b.z, b.w};
    return __builtin_bit_cast(v16bf, u);
}

#if HAVE_TDM
// LDS byte offset of a shared-memory object (AS3 pointers are 32-bit offsets)
static __device__ __forceinline__ unsigned lds_off(const void* p) {
    return (unsigned)(unsigned long long)(__attribute__((address_space(3))) const void*)p;
}
// Issue one 2D TDM tile load: global (bf16, row stride `stride0` elements)
// -> LDS packed [tile_d1][tile_d0].  Descriptor per CDNA5 ISA 08 §8.
static __device__ __forceinline__ void tdm_load_2d(
    const void* gaddr, unsigned lds_byte,
    unsigned tensor_d0, unsigned tensor_d1, unsigned stride0,
    unsigned tile_d0, unsigned tile_d1)
{
    const unsigned long long ga = (unsigned long long)(uintptr_t)gaddr;
    v4u g0;
    g0[0] = 1u;                                        // count=1 (valid), no gather
    g0[1] = lds_byte;                                  // lds_addr
    g0[2] = (unsigned)ga;                              // global_addr[31:0]
    g0[3] = (unsigned)(ga >> 32) | (2u << 30);         // global_addr[56:32] | type=2
    v8i g1;
    g1[0] = (int)(1u << 16);                           // data_size=1 -> 2 bytes/elem
    g1[1] = (int)((tensor_d0 & 0xFFFFu) << 16);        // tensor_dim0[15:0]
    g1[2] = (int)((tensor_d0 >> 16) | ((tensor_d1 & 0xFFFFu) << 16));
    g1[3] = (int)((tensor_d1 >> 16) | (tile_d0 << 16));// tensor_dim1[31:16] | tile_dim0
    g1[4] = (int)(tile_d1 & 0xFFFFu);                  // tile_dim1, tile_dim2=0
    g1[5] = (int)stride0;                              // tensor_dim0_stride[31:0]
    g1[6] = 0;                                         // stride0[47:32], dim1_stride lo
    g1[7] = 0;
    const v4i gz = {0, 0, 0, 0};
#if __clang_major__ >= 23
    const v8i gz8 = {0, 0, 0, 0, 0, 0, 0, 0};
    __builtin_amdgcn_tensor_load_to_lds(g0, g1, gz, gz, gz8, 0);
#else
    __builtin_amdgcn_tensor_load_to_lds(g0, g1, gz, gz, 0);
#endif
}
#endif  // HAVE_TDM

// ---------------------------------------------------------------------------
// Kernel 0a: X fp32 -> bf16 row-major.  8 elems / thread, fully coalesced.
// ---------------------------------------------------------------------------
__global__ __launch_bounds__(256) void cvt_x_kernel(
    const float* __restrict__ X, unsigned short* __restrict__ Xb)
{
    const size_t i = ((size_t)blockIdx.x * 256 + threadIdx.x) * 8;
    const float4 a = *(const float4*)&X[i];
    const float4 b = *(const float4*)&X[i + 4];
    uint4 o;
    o.x = pack2(f2bf(a.x), f2bf(a.y));
    o.y = pack2(f2bf(a.z), f2bf(a.w));
    o.z = pack2(f2bf(b.x), f2bf(b.y));
    o.w = pack2(f2bf(b.z), f2bf(b.w));
    *(uint4*)&Xb[i] = o;
}

// ---------------------------------------------------------------------------
// Kernel 0b: W fp32 [k][n] -> Wt bf16 [n][k] (transpose + convert).
// Thread handles 8 consecutive k for one n; 16-B packed store.
// ---------------------------------------------------------------------------
__global__ __launch_bounds__(256) void cvt_w_kernel(
    const float* __restrict__ W, unsigned short* __restrict__ Wt)
{
    const int flat = blockIdx.x * 256 + threadIdx.x;
    const int n  = flat >> 7;          // 0..3071
    const int k0 = (flat & 127) * 8;   // 0..1016
    unsigned short t[8];
#pragma unroll
    for (int j = 0; j < 8; ++j)
        t[j] = f2bf(W[(size_t)(k0 + j) * N3 + n]);
    uint4 o;
    o.x = pack2(t[0], t[1]); o.y = pack2(t[2], t[3]);
    o.z = pack2(t[4], t[5]); o.w = pack2(t[6], t[7]);
    *(uint4*)&Wt[(size_t)n * HID + k0] = o;
}

// ---------------------------------------------------------------------------
// Kernel 1: fused QKV projection, TDM double-buffered.
// Grid: (N3/128, (NBAT*SEQ)/16), block 256 (8 waves). Wave w owns cols
// [nb+16w, nb+16w+16). One WMMA per 32-deep K step.
// ---------------------------------------------------------------------------
__global__ __launch_bounds__(256) void qkv_proj_kernel(
    const unsigned short* __restrict__ Xb, const unsigned short* __restrict__ Wt,
    const float* __restrict__ bias,
    unsigned short* __restrict__ Qb, unsigned short* __restrict__ Kb,
    unsigned short* __restrict__ Vt)
{
    __shared__ alignas(16) unsigned short As[2][16][32];    // 2 x 1 KB
    __shared__ alignas(16) unsigned short Ws[2][128][32];   // 2 x 8 KB  ([n][k])

    const int tid  = threadIdx.x;
    const int lane = tid & 31;
    const int wave = tid >> 5;
    const int hi   = lane >> 4;
    const int m0   = blockIdx.y * 16;    // flat row (b*S + s)
    const int nb   = blockIdx.x * 128;   // block col base
    const int nn   = wave * 16 + (lane & 15);
    const int n    = nb + nn;

#if HAVE_TDM
    if (wave == 0) {  // prologue: prime buffer 0 (EXEC ignored by TDM; per-wave issue)
        tdm_load_2d(&Xb[(size_t)m0 * HID], lds_off(&As[0][0][0]),
                    HID, 16, HID, 32, 16);
        tdm_load_2d(&Wt[(size_t)nb * HID], lds_off(&Ws[0][0][0]),
                    HID, 128, HID, 32, 128);
    }
#endif

    v8f acc = {0.f, 0.f, 0.f, 0.f, 0.f, 0.f, 0.f, 0.f};

    for (int kb = 0, it = 0; kb < HID; kb += 32, ++it) {
        const int cur = it & 1;
        __syncthreads();               // all waves done reading buffer cur^1
#if HAVE_TDM
        if (wave == 0) {
            if (kb + 32 < HID) {       // prefetch next tile pair into cur^1
                tdm_load_2d(&Xb[(size_t)m0 * HID + (kb + 32)],
                            lds_off(&As[cur ^ 1][0][0]), HID, 16, HID, 32, 16);
                tdm_load_2d(&Wt[(size_t)nb * HID + (kb + 32)],
                            lds_off(&Ws[cur ^ 1][0][0]), HID, 128, HID, 32, 128);
                // 4 outstanding; <=2 left  <=>  current pair has landed
                __builtin_amdgcn_s_wait_tensorcnt(2);
            } else {
                __builtin_amdgcn_s_wait_tensorcnt(0);
            }
        }
#else
        {   // fallback cooperative staging (data already bf16)
            const int r = tid >> 4, wd = tid & 15;
            *(unsigned int*)&As[cur][r][wd * 2] =
                *(const unsigned int*)&Xb[(size_t)(m0 + r) * HID + kb + wd * 2];
            const int nr = tid >> 1, kh = (tid & 1) * 16;
            *(uint4*)&Ws[cur][nr][kh] =
                *(const uint4*)&Wt[(size_t)(nb + nr) * HID + kb + kh];
            *(uint4*)&Ws[cur][nr][kh + 8] =
                *(const uint4*)&Wt[(size_t)(nb + nr) * HID + kb + kh + 8];
        }
#endif
        __syncthreads();               // buffer cur is ready
        // A fragment: row = lane%16; K halves {hi*8..+7, hi*8+16..+23}
        const int arow = lane & 15;
        const uint4 a0 = *(const uint4*)&As[cur][arow][hi * 8];
        const uint4 a1 = *(const uint4*)&As[cur][arow][hi * 8 + 16];
        // B fragment: N = lane%16; K halves {hi*8..+7, hi*8+16..+23}
        const uint4 b0 = *(const uint4*)&Ws[cur][nn][hi * 8];
        const uint4 b1 = *(const uint4*)&Ws[cur][nn][hi * 8 + 16];
        acc = __builtin_amdgcn_wmma_f32_16x16x32_bf16(
            false, mk_frag(a0, a1), false, mk_frag(b0, b1),
            (short)0, acc, false, false);
    }

    // epilogue: bias add, route to Q / K / Vt (bf16)
    const float bv = bias[n];
    const int   b  = m0 / SEQ;
    const int   s0 = (m0 % SEQ) + hi * 8;     // 8 consecutive s per lane
    if (n < HID) {
#pragma unroll
        for (int r = 0; r < 8; ++r)
            Qb[((size_t)b * SEQ + s0 + r) * HID + n] = f2bf(acc[r] + bv);
    } else if (n < 2 * HID) {
#pragma unroll
        for (int r = 0; r < 8; ++r)
            Kb[((size_t)b * SEQ + s0 + r) * HID + (n - HID)] = f2bf(acc[r] + bv);
    } else {
        uint4 o;
        o.x = pack2(f2bf(acc[0] + bv), f2bf(acc[1] + bv));
        o.y = pack2(f2bf(acc[2] + bv), f2bf(acc[3] + bv));
        o.z = pack2(f2bf(acc[4] + bv), f2bf(acc[5] + bv));
        o.w = pack2(f2bf(acc[6] + bv), f2bf(acc[7] + bv));
        *(uint4*)&Vt[((size_t)b * HID + (n - 2 * HID)) * SEQ + s0] = o;  // contig in s
    }
}

// ---------------------------------------------------------------------------
// Kernel 2: streaming-softmax attention.
// Grid: (SEQ/16, NBAT), block 256 (8 waves). Workgroup owns 16 query rows;
// Q tile TDM-staged into LDS and resident. Per 128-key tile: wave w scores
// keys [kt+16w,+16); wave-0 lanes run the online softmax; each wave then
// accumulates its 128-wide d-slice of O = P @ V via transposed V.
// ---------------------------------------------------------------------------
__global__ __launch_bounds__(256) void attn_kernel(
    const unsigned short* __restrict__ Qb, const unsigned short* __restrict__ Kb,
    const unsigned short* __restrict__ Vt, float* __restrict__ out)
{
    __shared__ alignas(16) unsigned short Qs[16][HID];  // 32 KB resident Q tile
    __shared__ alignas(16) float          Ss[16][128];  // 8 KB scores
    __shared__ alignas(16) unsigned short Ps[16][128];  // 4 KB exp(scores) bf16
    __shared__ float m_s[16], l_s[16], c_s[16];

    const int tid  = threadIdx.x;
    const int lane = tid & 31;
    const int wave = tid >> 5;
    const int hi   = lane >> 4;
    const int b    = blockIdx.y;
    const int q0   = blockIdx.x * 16;

#if HAVE_TDM
    if (wave == 0) {   // one DMA: 16 x 1024 bf16 tile -> Qs
        tdm_load_2d(&Qb[((size_t)b * SEQ + q0) * HID], lds_off(&Qs[0][0]),
                    HID, 16, HID, HID, 16);
        __builtin_amdgcn_s_wait_tensorcnt(0);
    }
#else
    {
        const int r  = tid >> 4;
        const int c0 = (tid & 15) * 64;
        const uint4* src = (const uint4*)&Qb[((size_t)b * SEQ + q0 + r) * HID + c0];
        uint4* dst = (uint4*)&Qs[r][c0];
#pragma unroll
        for (int j = 0; j < 8; ++j) dst[j] = src[j];
    }
#endif
    if (tid < 16) { m_s[tid] = -__builtin_inff(); l_s[tid] = 0.f; }
    __syncthreads();

    const v8f vz = {0.f, 0.f, 0.f, 0.f, 0.f, 0.f, 0.f, 0.f};
    v8f acc[8];
#pragma unroll
    for (int nt = 0; nt < 8; ++nt) acc[nt] = vz;

    const float scale = 0.03125f;  // 1/sqrt(1024)

    for (int kt = 0; kt < SEQ; kt += 128) {
        // ---------- phase A: S = Q . K^T for this key tile ----------
        const int key = kt + wave * 16 + (lane & 15);
        const unsigned short* krow = &Kb[((size_t)b * SEQ + key) * HID];
        if (kt + 128 < SEQ)  // global_prefetch_b8 of next key tile
            __builtin_prefetch(&Kb[((size_t)b * SEQ + key + 128) * HID], 0, 1);

        v8f sc = vz;
        for (int d0 = 0; d0 < HID; d0 += 32) {
            const uint4 a0 = *(const uint4*)&Qs[lane & 15][d0 + hi * 8];
            const uint4 a1 = *(const uint4*)&Qs[lane & 15][d0 + hi * 8 + 16];
            const uint4 b0 = *(const uint4*)&krow[d0 + hi * 8];
            const uint4 b1 = *(const uint4*)&krow[d0 + hi * 8 + 16];
            sc = __builtin_amdgcn_wmma_f32_16x16x32_bf16(
                false, mk_frag(a0, a1), false, mk_frag(b0, b1),
                (short)0, sc, false, false);
        }
#pragma unroll
        for (int r = 0; r < 8; ++r)
            Ss[r + hi * 8][wave * 16 + (lane & 15)] = sc[r] * scale;
        __syncthreads();

        // ---------- phase B: online softmax update (one lane per row) ----------
        if (tid < 16) {
            const int row = tid;
            const float mold = m_s[row];
            float mx = mold;
            for (int j = 0; j < 128; j += 4) {
                const float4 s4 = *(const float4*)&Ss[row][j];
                mx = fmaxf(mx, fmaxf(fmaxf(s4.x, s4.y), fmaxf(s4.z, s4.w)));
            }
            const float cf = __expf(mold - mx);  // 0 on first tile (mold = -inf)
            float sum = 0.f;
            for (int j = 0; j < 128; j += 4) {
                const float4 s4 = *(const float4*)&Ss[row][j];
                const float e0 = __expf(s4.x - mx), e1 = __expf(s4.y - mx);
                const float e2 = __expf(s4.z - mx), e3 = __expf(s4.w - mx);
                sum += (e0 + e1) + (e2 + e3);
                uint2 p;
                p.x = pack2(f2bf(e0), f2bf(e1));
                p.y = pack2(f2bf(e2), f2bf(e3));
                *(uint2*)&Ps[row][j] = p;
            }
            m_s[row] = mx;
            c_s[row] = cf;
            l_s[row] = l_s[row] * cf + sum;
        }
        __syncthreads();

        // ---------- phase C: rescale + O += P . V (wave owns 128 dims) ----------
        float cr[8];
#pragma unroll
        for (int r = 0; r < 8; ++r) cr[r] = c_s[hi * 8 + r];
#pragma unroll
        for (int nt = 0; nt < 8; ++nt)
#pragma unroll
            for (int r = 0; r < 8; ++r) acc[nt][r] *= cr[r];

        for (int kc = 0; kc < 128; kc += 32) {
            const uint4 a0 = *(const uint4*)&Ps[lane & 15][kc + hi * 8];
            const uint4 a1 = *(const uint4*)&Ps[lane & 15][kc + hi * 8 + 16];
            const v16bf af = mk_frag(a0, a1);
            const int keyb = kt + kc + hi * 8;
#pragma unroll
            for (int nt = 0; nt < 8; ++nt) {
                const int d = wave * 128 + nt * 16 + (lane & 15);
                const unsigned short* vp = &Vt[((size_t)b * HID + d) * SEQ + keyb];
                const uint4 b0 = *(const uint4*)vp;
                const uint4 b1 = *(const uint4*)(vp + 16);
                acc[nt] = __builtin_amdgcn_wmma_f32_16x16x32_bf16(
                    false, af, false, mk_frag(b0, b1),
                    (short)0, acc[nt], false, false);
            }
        }
        __syncthreads();
    }

    // ---------- epilogue: O / l, fp32 store ----------
    float linv[8];
#pragma unroll
    for (int r = 0; r < 8; ++r) linv[r] = 1.f / l_s[hi * 8 + r];
#pragma unroll
    for (int nt = 0; nt < 8; ++nt) {
        const int d = wave * 128 + nt * 16 + (lane & 15);
#pragma unroll
        for (int r = 0; r < 8; ++r)
            out[((size_t)b * SEQ + q0 + hi * 8 + r) * HID + d] = acc[nt][r] * linv[r];
    }
}

// ---------------------------------------------------------------------------
extern "C" void kernel_launch(void* const* d_in, const int* in_sizes, int n_in,
                              void* d_out, int out_size, void* d_ws, size_t ws_size,
                              hipStream_t stream)
{
    const float* X    = (const float*)d_in[0];  // [4,2048,1024]
    const float* W    = (const float*)d_in[1];  // [1024,3072]
    const float* bias = (const float*)d_in[2];  // [3072]
    float* out        = (float*)d_out;          // [4,2048,1024]

    const size_t elems = (size_t)NBAT * SEQ * HID;       // 8 Mi
    unsigned short* Qb = (unsigned short*)d_ws;          // 16 MB
    unsigned short* Kb = Qb + elems;                     // 16 MB
    unsigned short* Vt = Kb + elems;                     // 16 MB (V transposed)
    unsigned short* Xb = Vt + elems;                     // 16 MB (bf16 X)
    unsigned short* Wt = Xb + elems;                     //  6 MB (bf16 W^T)
    (void)in_sizes; (void)n_in; (void)out_size; (void)ws_size;

    cvt_x_kernel<<<dim3(elems / (256 * 8)), 256, 0, stream>>>(X, Xb);
    cvt_w_kernel<<<dim3((N3 * (HID / 8)) / 256), 256, 0, stream>>>(W, Wt);

    dim3 g1(N3 / 128, (NBAT * SEQ) / 16);  // (24, 512)
    qkv_proj_kernel<<<g1, 256, 0, stream>>>(Xb, Wt, bias, Qb, Kb, Vt);

    dim3 g2(SEQ / 16, NBAT);               // (128, 4)
    attn_kernel<<<g2, 256, 0, stream>>>(Qb, Kb, Vt, out);
}
